// GCN_11175504904923
// MI455X (gfx1250) — matrix operand order repaired and checked by reading, not compiled, not run on hardware
//
#include <hip/hip_runtime.h>
#include <hip/hip_bf16.h>
#include <math.h>

// ---- problem constants (match reference) ----
#define FEAT    128
#define HID     16
#define NCLS    2
#define NGRAPH  512

typedef __attribute__((ext_vector_type(16))) _Float16 v16h;
typedef __attribute__((ext_vector_type(8)))  float    v8f;

// ---------------- degree / normalization ----------------
__global__ void k_init_deg(float* __restrict__ deg, int n) {
    int i = blockIdx.x * blockDim.x + threadIdx.x;
    if (i < n) deg[i] = 1.0f;                       // self-loop contributes 1
}

__global__ void k_deg_accum(const int* __restrict__ dst, float* __restrict__ deg, int e) {
    int i = blockIdx.x * blockDim.x + threadIdx.x;
    if (i < e) atomicAdd(&deg[dst[i]], 1.0f);
}

__global__ void k_dinv(float* __restrict__ deg, int n) {
    int i = blockIdx.x * blockDim.x + threadIdx.x;
    if (i < n) {
        float d = deg[i];
        deg[i] = (d > 0.0f) ? rsqrtf(d) : 0.0f;     // deg >= 1 always (self-loops)
    }
}

// ---------------- layer 1 GEMM: h1 = x @ W1 via WMMA ----------------
// One wave computes a 16(node)x16(hidden) f32 tile; K=128 -> 4 wmma f16 steps.
__global__ void __launch_bounds__(256) k_gemm1_wmma(const float* __restrict__ x,
                                                    const float* __restrict__ W1,
                                                    float* __restrict__ h1, int n) {
    __shared__ float ws[FEAT * HID];                // 8 KB: whole W1 staged in LDS
    for (int t = threadIdx.x; t < FEAT * HID; t += blockDim.x) ws[t] = W1[t];
    __syncthreads();

    const int lane = threadIdx.x & 31;
    const int wave = threadIdx.x >> 5;
    const int tile = blockIdx.x * 8 + wave;         // 8 waves per block
    const int row0 = tile * 16;
    if (row0 >= n) return;

    const int col  = lane & 15;                     // N index (and M sub-row for A)
    const int half = lane >> 4;                     // which K-half this lane owns
    const int row  = row0 + col;                    // A-matrix row for this lane

    v8f acc = {};
    const float* xr = x + (size_t)row * FEAT + half * 8;

#pragma unroll
    for (int kb = 0; kb < FEAT; kb += 32) {
        // ---- A fragment: 16x32 f16. lane<16: K in {kb..kb+7, kb+16..kb+23};
        //      lane>=16: shifted by +8. 16B-aligned float4 global loads.
        float4 f0 = *(const float4*)(xr + kb);
        float4 f1 = *(const float4*)(xr + kb + 4);
        float4 f2 = *(const float4*)(xr + kb + 16);
        float4 f3 = *(const float4*)(xr + kb + 20);
        v16h a;
        a[0]  = (_Float16)f0.x; a[1]  = (_Float16)f0.y; a[2]  = (_Float16)f0.z; a[3]  = (_Float16)f0.w;
        a[4]  = (_Float16)f1.x; a[5]  = (_Float16)f1.y; a[6]  = (_Float16)f1.z; a[7]  = (_Float16)f1.w;
        a[8]  = (_Float16)f2.x; a[9]  = (_Float16)f2.y; a[10] = (_Float16)f2.z; a[11] = (_Float16)f2.w;
        a[12] = (_Float16)f3.x; a[13] = (_Float16)f3.y; a[14] = (_Float16)f3.z; a[15] = (_Float16)f3.w;

        // ---- B fragment: 32x16 f16, column-per-lane, K split by half-wave (LDS reads)
        const int kbase = kb + half * 8;
        v16h b;
#pragma unroll
        for (int j = 0; j < 16; ++j)
            b[j] = (_Float16)ws[(kbase + j + (j & 8)) * HID + col];

        acc = __builtin_amdgcn_wmma_f32_16x16x32_f16(
            /*neg_a=*/false, a, /*neg_b=*/false, b,
            /*c_mod=*/(short)0, acc, /*reuse_a=*/false, /*reuse_b=*/false);
    }

    // ---- store C: VGPR v holds M = v + 8*half, N = col
    float* out = h1 + (size_t)row0 * HID;
#pragma unroll
    for (int v = 0; v < 8; ++v)
        out[(size_t)(v + half * 8) * HID + col] = acc[v];
}

// ---------------- agg1 init: bias + self-loop term (float4 vectorized) ----------------
__global__ void k_init_agg1(const float* __restrict__ h1, const float* __restrict__ dinv,
                            const float* __restrict__ b1, float* __restrict__ agg1, int n) {
    int i = blockIdx.x * blockDim.x + threadIdx.x;  // over n*4 quarters of a row
    if (i >= n * 4) return;
    int node = i >> 2, q = (i & 3) * 4;
    float di = dinv[node];
    float di2 = di * di;
    float4 hv = *(const float4*)(h1 + (size_t)node * HID + q);
    float4 bv = *(const float4*)(b1 + q);
    float4 o;
    o.x = bv.x + hv.x * di2;
    o.y = bv.y + hv.y * di2;
    o.z = bv.z + hv.z * di2;
    o.w = bv.w + hv.w * di2;
    *(float4*)(agg1 + (size_t)node * HID + q) = o;
}

// ---------------- layer-1 edge scatter: one thread per edge ----------------
// Indices and norm loaded once per edge (saves ~460 MB vs 4-threads/edge),
// four b128 gathers of h1, sixteen global_atomic_add_f32.
__global__ void k_scatter1(const int* __restrict__ src, const int* __restrict__ dst,
                           const float* __restrict__ dinv, const float* __restrict__ h1,
                           float* __restrict__ agg1, int e) {
    int ed = blockIdx.x * blockDim.x + threadIdx.x;
    if (ed >= e) return;
    int s = src[ed], d = dst[ed];
    float w = dinv[s] * dinv[d];
    const float4* hp = (const float4*)(h1 + (size_t)s * HID);
    float* ap = agg1 + (size_t)d * HID;
#pragma unroll
    for (int q = 0; q < 4; ++q) {
        float4 hv = hp[q];
        atomicAdd(ap + q * 4 + 0, hv.x * w);
        atomicAdd(ap + q * 4 + 1, hv.y * w);
        atomicAdd(ap + q * 4 + 2, hv.z * w);
        atomicAdd(ap + q * 4 + 3, hv.w * w);
    }
}

// ---------------- layer 2: h2 = relu(agg1) @ W2 (16 -> 2) ----------------
__global__ void k_gemm2_relu(const float* __restrict__ agg1, const float* __restrict__ W2,
                             float* __restrict__ h2, int n) {
    int i = blockIdx.x * blockDim.x + threadIdx.x;
    if (i >= n) return;
    const float4* hp = (const float4*)(agg1 + (size_t)i * HID);
    float o0 = 0.0f, o1 = 0.0f;
#pragma unroll
    for (int q = 0; q < 4; ++q) {
        float4 hv = hp[q];
        float v0 = fmaxf(hv.x, 0.0f), v1 = fmaxf(hv.y, 0.0f);
        float v2 = fmaxf(hv.z, 0.0f), v3 = fmaxf(hv.w, 0.0f);
        int f = q * 4;
        o0 += v0 * W2[(f + 0) * NCLS] + v1 * W2[(f + 1) * NCLS]
            + v2 * W2[(f + 2) * NCLS] + v3 * W2[(f + 3) * NCLS];
        o1 += v0 * W2[(f + 0) * NCLS + 1] + v1 * W2[(f + 1) * NCLS + 1]
            + v2 * W2[(f + 2) * NCLS + 1] + v3 * W2[(f + 3) * NCLS + 1];
    }
    h2[(size_t)i * NCLS + 0] = o0;
    h2[(size_t)i * NCLS + 1] = o1;
}

__global__ void k_init_agg2(const float* __restrict__ h2, const float* __restrict__ dinv,
                            const float* __restrict__ b2, float* __restrict__ agg2, int n) {
    int i = blockIdx.x * blockDim.x + threadIdx.x;
    if (i >= n) return;
    float di2 = dinv[i] * dinv[i];
    agg2[(size_t)i * NCLS + 0] = b2[0] + h2[(size_t)i * NCLS + 0] * di2;
    agg2[(size_t)i * NCLS + 1] = b2[1] + h2[(size_t)i * NCLS + 1] * di2;
}

__global__ void k_scatter2(const int* __restrict__ src, const int* __restrict__ dst,
                           const float* __restrict__ dinv, const float* __restrict__ h2,
                           float* __restrict__ agg2, int e) {
    int ed = blockIdx.x * blockDim.x + threadIdx.x;
    if (ed >= e) return;
    int s = src[ed], d = dst[ed];
    float w = dinv[s] * dinv[d];
    float2 hv = *(const float2*)(h2 + (size_t)s * NCLS);
    atomicAdd(&agg2[(size_t)d * NCLS + 0], hv.x * w);
    atomicAdd(&agg2[(size_t)d * NCLS + 1], hv.y * w);
}

// ---------------- pooling + log-softmax ----------------
__global__ void k_pool_init(float* __restrict__ pooled, float* __restrict__ counts) {
    int i = blockIdx.x * blockDim.x + threadIdx.x;
    if (i < NGRAPH * NCLS) pooled[i] = 0.0f;
    if (i < NGRAPH) counts[i] = 0.0f;
}

__global__ void k_pool(const int* __restrict__ batch, const float* __restrict__ agg2,
                       float* __restrict__ pooled, float* __restrict__ counts, int n) {
    int i = blockIdx.x * blockDim.x + threadIdx.x;
    if (i >= n) return;
    int g = batch[i];
    float2 av = *(const float2*)(agg2 + (size_t)i * NCLS);
    atomicAdd(&pooled[g * NCLS + 0], av.x);
    atomicAdd(&pooled[g * NCLS + 1], av.y);
    atomicAdd(&counts[g], 1.0f);
}

__global__ void k_final(const float* __restrict__ pooled, const float* __restrict__ counts,
                        float* __restrict__ out) {
    int g = blockIdx.x * blockDim.x + threadIdx.x;
    if (g >= NGRAPH) return;
    float c  = fmaxf(counts[g], 1.0f);
    float z0 = pooled[g * NCLS + 0] / c;
    float z1 = pooled[g * NCLS + 1] / c;
    float m  = fmaxf(z0, z1);
    float lse = m + logf(expf(z0 - m) + expf(z1 - m));
    out[g * NCLS + 0] = z0 - lse;
    out[g * NCLS + 1] = z1 - lse;
}

// ---------------- launch ----------------
extern "C" void kernel_launch(void* const* d_in, const int* in_sizes, int n_in,
                              void* d_out, int out_size, void* d_ws, size_t ws_size,
                              hipStream_t stream) {
    const float* x    = (const float*)d_in[0];
    const int*   ei   = (const int*)d_in[1];     // [2, E] flattened: src then dst
    const int*   bat  = (const int*)d_in[2];
    const float* W1   = (const float*)d_in[3];
    const float* b1   = (const float*)d_in[4];
    const float* W2   = (const float*)d_in[5];
    const float* b2   = (const float*)d_in[6];
    float*       out  = (float*)d_out;

    const int n = in_sizes[0] / FEAT;            // 300000
    const int e = in_sizes[1] / 2;               // 9600000
    const int* src = ei;
    const int* dst = ei + e;

    // workspace layout (floats)
    float* dinv   = (float*)d_ws;                // n   (deg -> dinv in place)
    float* h1     = dinv   + n;                  // n*16
    float* agg1   = h1     + (size_t)n * HID;    // n*16
    float* h2     = agg1   + (size_t)n * HID;    // n*2
    float* agg2   = h2     + (size_t)n * NCLS;   // n*2
    float* pooled = agg2   + (size_t)n * NCLS;   // 1024
    float* counts = pooled + NGRAPH * NCLS;      // 512

    const int T = 256;
    auto blocks = [](long long total, int t) { return (unsigned)((total + t - 1) / t); };

    // 1) degree / D^{-1/2}
    k_init_deg <<<blocks(n, T), T, 0, stream>>>(dinv, n);
    k_deg_accum<<<blocks(e, T), T, 0, stream>>>(dst, dinv, e);
    k_dinv     <<<blocks(n, T), T, 0, stream>>>(dinv, n);

    // 2) h1 = x @ W1 (WMMA, 8 tiles of 16 rows per block)
    {
        int tiles = (n + 15) / 16;
        k_gemm1_wmma<<<blocks(tiles, 8), 256, 0, stream>>>(x, W1, h1, n);
    }

    // 3) layer-1 aggregation
    k_init_agg1<<<blocks((long long)n * 4, T), T, 0, stream>>>(h1, dinv, b1, agg1, n);
    k_scatter1 <<<blocks(e, T), T, 0, stream>>>(src, dst, dinv, h1, agg1, e);

    // 4) layer 2 transform + aggregation
    k_gemm2_relu<<<blocks(n, T), T, 0, stream>>>(agg1, W2, h2, n);
    k_init_agg2 <<<blocks(n, T), T, 0, stream>>>(h2, dinv, b2, agg2, n);
    k_scatter2  <<<blocks(e, T), T, 0, stream>>>(src, dst, dinv, h2, agg2, e);

    // 5) mean pool + log_softmax
    k_pool_init<<<blocks(NGRAPH * NCLS, T), T, 0, stream>>>(pooled, counts);
    k_pool     <<<blocks(n, T), T, 0, stream>>>(bat, agg2, pooled, counts, n);
    k_final    <<<blocks(NGRAPH, T), T, 0, stream>>>(pooled, counts, out);
}
